// SimpleMessagePassingGNN_12120397710063
// MI455X (gfx1250) — compile-verified, matrix operand order checked by reading, not run on hardware
//
#include <hip/hip_runtime.h>
#include <hip/hip_bf16.h>

typedef __attribute__((ext_vector_type(2)))  float     v2f;
typedef __attribute__((ext_vector_type(8)))  float     v8f;
typedef __attribute__((ext_vector_type(16))) _Float16  v16h;

#define CH 128   // IN_CH == OUT_CH == 128

// ---------------------------------------------------------------------------
// Zero workspace (S [N x 128] and deg [N] are contiguous)
// ---------------------------------------------------------------------------
__global__ __launch_bounds__(256) void zero_f32(float* __restrict__ p, long n) {
    long i = (long)blockIdx.x * blockDim.x + threadIdx.x;
    long stride = (long)gridDim.x * blockDim.x;
    for (; i < n; i += stride) p[i] = 0.0f;
}

// ---------------------------------------------------------------------------
// Wc = W_upd @ W_msg  (128x128), bc = W_upd @ b_msg  (128)
// One block per output row o, one thread per column k.
// ---------------------------------------------------------------------------
__global__ __launch_bounds__(CH) void fuse_weights(const float* __restrict__ Wu,
                                                   const float* __restrict__ Wm,
                                                   const float* __restrict__ bm,
                                                   float* __restrict__ Wc,
                                                   float* __restrict__ bc) {
    const int o = blockIdx.x;
    const int k = threadIdx.x;
    float acc = 0.0f;
    for (int j = 0; j < CH; ++j) acc += Wu[o * CH + j] * Wm[j * CH + k];
    Wc[o * CH + k] = acc;
    if (k == 0) {
        float a = 0.0f;
        for (int j = 0; j < CH; ++j) a += Wu[o * CH + j] * bm[j];
        bc[o] = a;
    }
}

// ---------------------------------------------------------------------------
// Edge scatter: one edge per wave32. Lanes cover 128 channels (4 per lane,
// stride-32 for coalesced 128B bursts). Atomic f32 adds into S[tgt], plus a
// degree counter (needed for the deg[t]*bc bias term).
// ---------------------------------------------------------------------------
__global__ __launch_bounds__(256) void edge_scatter(const float* __restrict__ x,
                                                    const int* __restrict__ ei,
                                                    int E,
                                                    float* __restrict__ S,
                                                    float* __restrict__ deg) {
    const int lane = threadIdx.x & 31;
    const long w = (long)blockIdx.x * 8 + (threadIdx.x >> 5);
    if (w >= E) return;
    const int src = ei[w];
    const int tgt = ei[(long)E + w];
    const float* __restrict__ xr = x + (long)src * CH;
    float* __restrict__ sr = S + (long)tgt * CH;
#pragma unroll
    for (int i = 0; i < 4; ++i) {
        const int c = lane + 32 * i;
        atomicAdd(&sr[c], xr[c]);
    }
    if (lane == 0) atomicAdd(&deg[tgt], 1.0f);
}

// ---------------------------------------------------------------------------
// out = S @ Wc^T + deg (x) bc + b_upd
// 256 threads = 8 wave32; wave w owns a 16-node x 16-outch tile (cols 16w..).
// Preferred path: native f32 WMMA 16x16x4 (K-steps of 4, 32 per tile).
// Fallback: f16 WMMA 16x16x32 with f32 accumulate (codegen-confirmed).
// Row loads are clamped so EXEC stays all-ones through the WMMA loop.
// ---------------------------------------------------------------------------
__global__ __launch_bounds__(256) void node_gemm(const float* __restrict__ S,
                                                 const float* __restrict__ deg,
                                                 const float* __restrict__ Wc,
                                                 const float* __restrict__ bc,
                                                 const float* __restrict__ bu,
                                                 float* __restrict__ out,
                                                 int N) {
    const int lane = threadIdx.x & 31;
    const int wave = threadIdx.x >> 5;
    const int nb = blockIdx.x * 16;   // node tile base (rows, M)
    const int cb = wave * 16;         // out-channel tile base (cols, N)
    const int half = lane >> 4;       // 0: lanes 0-15, 1: lanes 16-31
    const int ml = lane & 15;

    int arow = nb + ml;
    if (arow > N - 1) arow = N - 1;   // clamp: keep EXEC all-ones for WMMA
    const float* __restrict__ sRow = S + (long)arow * CH;        // A row (M=ml)
    const float* __restrict__ wRow = Wc + (long)(cb + ml) * CH;  // B col (N=ml)

    v8f c = {};

#if defined(__has_builtin) && __has_builtin(__builtin_amdgcn_wmma_f32_16x16x4_f32)
    // A 16x4 f32 layout: lane(0-15)->M=lane, VGPR0/1 = K0/K1; lanes 16-31 = K2/K3.
    // B 4x16 mirrors it with N striped across lanes.  K advances by 4 per step.
#pragma unroll
    for (int step = 0; step < CH / 4; ++step) {
        const int k0 = step * 4 + 2 * half;
        v2f a = *reinterpret_cast<const v2f*>(sRow + k0);
        v2f b = *reinterpret_cast<const v2f*>(wRow + k0);
        c = __builtin_amdgcn_wmma_f32_16x16x4_f32(
                false, a, false, b, (short)0, c, false, false);
    }
#else
    // f16 path: A 16x32 layout -> lane holds K = 8*half + {0..7} and +16..+23.
    const int base = 8 * half;
#pragma unroll
    for (int step = 0; step < CH / 32; ++step) {
        const int k0 = step * 32;
        v16h a, b;
#pragma unroll
        for (int t = 0; t < 8; ++t) {
            a[t]     = (_Float16)sRow[k0 + base + t];
            a[8 + t] = (_Float16)sRow[k0 + base + 16 + t];
            b[t]     = (_Float16)wRow[k0 + base + t];
            b[8 + t] = (_Float16)wRow[k0 + base + 16 + t];
        }
        c = __builtin_amdgcn_wmma_f32_16x16x32_f16(
                false, a, false, b, (short)0, c, false, false);
    }
#endif

    // Epilogue: D VGPR r -> M = r + 8*half, N = ml (ISA 16x16 f32 C/D layout).
    const int col = cb + ml;
    const float bcv = bc[col];
    const float buv = bu[col];
#pragma unroll
    for (int r = 0; r < 8; ++r) {
        const int row = nb + r + 8 * half;
        if (row < N) {
            out[(long)row * CH + col] = c[r] + deg[row] * bcv + buv;
        }
    }
}

// ---------------------------------------------------------------------------
extern "C" void kernel_launch(void* const* d_in, const int* in_sizes, int n_in,
                              void* d_out, int out_size, void* d_ws, size_t ws_size,
                              hipStream_t stream) {
    const float* x      = (const float*)d_in[0];   // [N, 128]
    const int*   ei     = (const int*)d_in[1];     // [2, E]
    const float* W_msg  = (const float*)d_in[2];   // [128, 128]
    const float* b_msg  = (const float*)d_in[3];   // [128]
    const float* W_upd  = (const float*)d_in[4];   // [128, 128]
    const float* b_upd  = (const float*)d_in[5];   // [128]
    float*       out    = (float*)d_out;           // [N, 128]

    const int N = in_sizes[0] / CH;
    const int E = in_sizes[1] / 2;

    // Workspace carve: S [N*128] | deg [N] | Wc [128*128] | bc [128]
    float* S   = (float*)d_ws;
    float* deg = S + (long)N * CH;
    float* Wc  = deg + N;
    float* bc  = Wc + CH * CH;

    // 1) zero S + deg (contiguous region)
    zero_f32<<<1024, 256, 0, stream>>>(S, (long)N * CH + N);

    // 2) fuse weights: Wc = Wu @ Wm, bc = Wu @ b_msg
    fuse_weights<<<CH, CH, 0, stream>>>(W_upd, W_msg, b_msg, Wc, bc);

    // 3) scatter-add raw features over edges (1 edge per wave32)
    edge_scatter<<<(E + 7) / 8, 256, 0, stream>>>(x, ei, E, S, deg);

    // 4) fused node GEMM + bias epilogue via WMMA
    node_gemm<<<(N + 15) / 16, 256, 0, stream>>>(S, deg, Wc, bc, b_upd, out, N);
}
